// LayerHead_10256381903693
// MI455X (gfx1250) — matrix operand order repaired and checked by reading, not compiled, not run on hardware
//
#include <hip/hip_runtime.h>

// Problem constants
#define Bq    2
#define Sq    4096
#define Hq    2048
#define NHq   32
#define NKVq  8
#define Dq    64
#define NTOT  3072   // H + KV + KV = 2048 + 512 + 512

typedef __bf16 bf16;
typedef __attribute__((ext_vector_type(16))) __bf16 v16bf;
typedef __attribute__((ext_vector_type(8)))  __bf16 v8bf;
typedef __attribute__((ext_vector_type(8)))  float  v8f;

union V16U { v16bf v; v8bf h[2]; };

// Output layout (flat f32, concatenated in return order)
#define RES_ELE  ((size_t)Bq * Sq * Hq)                  // 16,777,216
#define Q_OFF    (RES_ELE)                               // +B*NH*S*D
#define K_OFF    (Q_OFF + (size_t)Bq * NHq * Sq * Dq)    // +B*NKV*S*D
#define V_OFF    (K_OFF + (size_t)Bq * NKVq * Sq * Dq)

// ---------------------------------------------------------------------------
// Kernel 1: pack qW/kW/vW into a single (NTOT x H) bf16 weight matrix + bias
// ---------------------------------------------------------------------------
__global__ void pack_weights_kernel(const float* __restrict__ qW, const float* __restrict__ qb,
                                    const float* __restrict__ kW, const float* __restrict__ kb,
                                    const float* __restrict__ vW, const float* __restrict__ vb,
                                    bf16* __restrict__ Wc, float* __restrict__ biasc) {
  const size_t idx = (size_t)blockIdx.x * blockDim.x + threadIdx.x;
  if (idx < (size_t)NTOT * Hq) {
    const int row = (int)(idx / Hq);
    const int col = (int)(idx % Hq);
    float val;
    if (row < Hq)            val = qW[(size_t)row * Hq + col];
    else if (row < Hq + 512) val = kW[(size_t)(row - Hq) * Hq + col];
    else                     val = vW[(size_t)(row - Hq - 512) * Hq + col];
    Wc[idx] = (bf16)val;
  }
  if (idx < NTOT) {
    const int r = (int)idx;
    biasc[r] = (r < Hq) ? qb[r] : (r < Hq + 512 ? kb[r - Hq] : vb[r - Hq - 512]);
  }
}

// ---------------------------------------------------------------------------
// Kernel 2: LayerNorm (f32) + cast to bf16; also emits the residual output.
// One 256-thread block per row, 8 elements per thread.
// ---------------------------------------------------------------------------
__global__ void ln_cast_kernel(const float* __restrict__ x,
                               const float* __restrict__ w,
                               const float* __restrict__ bparm,
                               float* __restrict__ resid,
                               bf16* __restrict__ hbf) {
  __shared__ float r1[256], r2[256];
  const int row = blockIdx.x;
  const int t = threadIdx.x;
  const float* xr = x + (size_t)row * Hq;

  float v[8];
  float s1 = 0.f, s2 = 0.f;
#pragma unroll
  for (int i = 0; i < 8; ++i) {
    v[i] = xr[t + 256 * i];
    s1 += v[i];
    s2 += v[i] * v[i];
  }
  r1[t] = s1; r2[t] = s2;
  __syncthreads();
  for (int off = 128; off > 0; off >>= 1) {
    if (t < off) { r1[t] += r1[t + off]; r2[t] += r2[t + off]; }
    __syncthreads();
  }
  const float mu  = r1[0] * (1.0f / Hq);
  const float var = r2[0] * (1.0f / Hq) - mu * mu;
  const float rs  = rsqrtf(var + 1e-5f);
#pragma unroll
  for (int i = 0; i < 8; ++i) {
    const int c = t + 256 * i;
    resid[(size_t)row * Hq + c] = v[i];
    const float hh = (v[i] - mu) * rs * w[c] + bparm[c];
    hbf[(size_t)row * Hq + c] = (bf16)hh;
  }
}

// ---------------------------------------------------------------------------
// Kernel 3: bf16 WMMA GEMM (8192x3072x2048) + bias + fused RoPE + transpose.
// Block = 128 threads = 4 waves in a 2x2 grid; each wave owns a 64x64 tile
// (4x4 fragments of v_wmma_f32_16x16x32_bf16). A wave's 64-aligned, 64-wide
// N-range covers exactly one head, so rotate_half is intra-lane (tile j<->j^2).
// ---------------------------------------------------------------------------
__global__ __launch_bounds__(128)
void qkv_wmma_kernel(const bf16* __restrict__ hbf,
                     const bf16* __restrict__ Wc,
                     const float* __restrict__ biasc,
                     const float* __restrict__ cosp,
                     const float* __restrict__ sinp,
                     float* __restrict__ out) {
  const int lane = threadIdx.x & 31;
  const int wv   = threadIdx.x >> 5;
  const int lp   = lane & 15;   // row/col within 16
  const int half = lane >> 4;   // lane half selects K sub-range
  const int m0 = blockIdx.x * 128 + (wv >> 1) * 64;
  const int n0 = blockIdx.y * 128 + (wv & 1) * 64;

  v8f acc[4][4] = {};

  for (int k0 = 0; k0 < Hq; k0 += 32) {
    V16U  a[4];
    v16bf bm[4];
    // A fragments: 16x32 bf16, lane lp = row, lanes 0-15 hold K=0..15,
    // lanes 16-31 hold K=16..31 (contiguous, matching B).
#pragma unroll
    for (int i = 0; i < 4; ++i) {
      const bf16* ap = hbf + (size_t)(m0 + i * 16 + lp) * Hq + k0 + half * 16;
      a[i].h[0] = *(const v8bf*)(ap);
      a[i].h[1] = *(const v8bf*)(ap + 8);
    }
    // B fragments: 32x16 bf16, lane lp = column; lanes 0-15 hold K=0..15,
    // lanes 16-31 hold K=16..31 (contiguous run of W's N-major row).
#pragma unroll
    for (int j = 0; j < 4; ++j) {
      const bf16* bp = Wc + (size_t)(n0 + j * 16 + lp) * Hq + k0 + half * 16;
      bm[j] = *(const v16bf*)(bp);
    }
    if (k0 + 32 < Hq) {
      __builtin_prefetch(hbf + (size_t)(m0 + lp) * Hq + (k0 + 32), 0, 3);
      __builtin_prefetch(Wc  + (size_t)(n0 + lp) * Hq + (k0 + 32), 0, 3);
    }
#pragma unroll
    for (int i = 0; i < 4; ++i)
#pragma unroll
      for (int j = 0; j < 4; ++j)
        acc[i][j] = __builtin_amdgcn_wmma_f32_16x16x32_bf16(
            false, a[i].v, false, bm[j], (short)0, acc[i][j], false, false);
  }

  // ---- epilogue: bias + RoPE + scatter (wave-uniform region) ----
  const int region = (n0 < Hq) ? 0 : (n0 < Hq + 512 ? 1 : 2);  // q / k / v
  float cb[4];
#pragma unroll
  for (int j = 0; j < 4; ++j) cb[j] = biasc[n0 + j * 16 + lp];

#pragma unroll
  for (int i = 0; i < 4; ++i) {
    float tb[4][8];
#pragma unroll
    for (int j = 0; j < 4; ++j)
#pragma unroll
      for (int r = 0; r < 8; ++r)
        tb[j][r] = acc[i][j][r] + cb[j];

#pragma unroll
    for (int j = 0; j < 4; ++j) {
      const int n = n0 + j * 16 + lp;
#pragma unroll
      for (int r = 0; r < 8; ++r) {
        // C/D layout: lanes 0-15 -> M = r, lanes 16-31 -> M = 8 + r
        const int m  = m0 + i * 16 + half * 8 + r;
        const int bb = m >> 12;        // m / S
        const int ss = m & (Sq - 1);   // m % S
        float val = tb[j][r];
        if (region < 2) {              // RoPE for q and k
          const int d = n & 63;
          const float cv = cosp[(size_t)m * Dq + d];
          const float sv = sinp[(size_t)m * Dq + d];
          // rotate_half: d<32 -> -x[d+32] (tile j+2); d>=32 -> +x[d-32]
          const float rotv = (j < 2) ? -tb[j + 2][r] : tb[j - 2][r];
          val = val * cv + rotv * sv;
        }
        size_t off;
        if (region == 0) {
          const int hd = n >> 6, d = n & 63;
          off = Q_OFF + (((size_t)(bb * NHq + hd)) * Sq + ss) * Dq + d;
        } else if (region == 1) {
          const int nn = n - Hq, hd = nn >> 6, d = nn & 63;
          off = K_OFF + (((size_t)(bb * NKVq + hd)) * Sq + ss) * Dq + d;
        } else {
          const int nn = n - (Hq + 512), hd = nn >> 6, d = nn & 63;
          off = V_OFF + (((size_t)(bb * NKVq + hd)) * Sq + ss) * Dq + d;
        }
        out[off] = val;
      }
    }
  }
}

// ---------------------------------------------------------------------------
extern "C" void kernel_launch(void* const* d_in, const int* in_sizes, int n_in,
                              void* d_out, int out_size, void* d_ws, size_t ws_size,
                              hipStream_t stream) {
  const float* hidden = (const float*)d_in[0];
  const float* cosp   = (const float*)d_in[1];
  const float* sinp   = (const float*)d_in[2];
  const float* ln_w   = (const float*)d_in[3];
  const float* ln_b   = (const float*)d_in[4];
  const float* qW     = (const float*)d_in[5];
  const float* qb     = (const float*)d_in[6];
  const float* kW     = (const float*)d_in[7];
  const float* kb     = (const float*)d_in[8];
  const float* vW     = (const float*)d_in[9];
  const float* vb     = (const float*)d_in[10];
  float* out = (float*)d_out;

  // Workspace layout: [hbf: B*S*H bf16][Wc: NTOT*H bf16][biasc: NTOT f32]
  bf16*  hbf   = (bf16*)d_ws;
  bf16*  Wc    = (bf16*)((char*)d_ws + RES_ELE * 2);
  float* biasc = (float*)((char*)d_ws + RES_ELE * 2 + (size_t)NTOT * Hq * 2);

  pack_weights_kernel<<<((size_t)NTOT * Hq + 255) / 256, 256, 0, stream>>>(
      qW, qb, kW, kb, vW, vb, Wc, biasc);
  ln_cast_kernel<<<Bq * Sq, 256, 0, stream>>>(hidden, ln_w, ln_b, out, hbf);
  qkv_wmma_kernel<<<dim3((Bq * Sq) / 128, NTOT / 128), 128, 0, stream>>>(
      hbf, Wc, biasc, cosp, sinp, out);
}
